// NonLocalBlock_4655744548983
// MI455X (gfx1250) — matrix operand order repaired and checked by reading, not compiled, not run on hardware
//
#include <hip/hip_runtime.h>

// ---------------------------------------------------------------------------
// NonLocalBlock fused attention for gfx1250 (MI455X), wave32 + bf16 WMMA.
//   B=2, C=64, N=20*20*20=8000
//   sim = (phi^T theta)/8 ; att = softmax over FIRST N axis (per-column m)
//   out = gamma * W_mask @ (att-weighted g) + x
// Two N^2 GEMM sweeps in bf16 WMMA (f32 accum); channel-mixing projections in
// exact f32 VALU (<2% of FLOPs). ~15MB of intermediates live in L2; the 512MB
// attention matrix is never materialized. Per-wave 2x tiling halves L2 reads;
// online-softmax halves are merged per lane-half so the hot loops contain no
// cross-lane DS ops.
// ---------------------------------------------------------------------------

#define Bb 2
#define Cc 64
#define Nn 8000

typedef __attribute__((ext_vector_type(16))) __bf16 v16bf;
typedef __attribute__((ext_vector_type(8)))  __bf16 v8bf;
typedef __attribute__((ext_vector_type(8)))  float  v8f;

__device__ __forceinline__ v8f wmma_bf16(v16bf a, v16bf b, v8f c) {
  // 8 args: (neg_a, A, neg_b, B, c_mod, C, reuse_a, reuse_b)
  return __builtin_amdgcn_wmma_f32_16x16x32_bf16(false, a, false, b,
                                                 (short)0, c, false, false);
}

// contiguous 16 bf16 (two 16B loads, assembled)
__device__ __forceinline__ v16bf load16(const __bf16* p) {
  v8bf lo = *(const v8bf*)p;
  v8bf hi = *(const v8bf*)(p + 8);
  v16bf r;
#pragma unroll
  for (int i = 0; i < 8; ++i) { r[i] = lo[i]; r[i + 8] = hi[i]; }
  return r;
}

// A-fragment halves live at two non-contiguous 16B spans
__device__ __forceinline__ v16bf load16_split(const __bf16* p0, const __bf16* p1) {
  v8bf lo = *(const v8bf*)p0;
  v8bf hi = *(const v8bf*)p1;
  v16bf r;
#pragma unroll
  for (int i = 0; i < 8; ++i) { r[i] = lo[i]; r[i + 8] = hi[i]; }
  return r;
}

// ---------------------------------------------------------------------------
// Phase A: phi/theta/g projections. phiT/thetaT stored [B][N][C] bf16 so that
// WMMA fragments are contiguous per-lane loads; g stored [B][C][N] f32.
// ---------------------------------------------------------------------------
__global__ __launch_bounds__(256)
void proj_kernel(const float* __restrict__ x,
                 const float* __restrict__ wphi,
                 const float* __restrict__ wtheta,
                 const float* __restrict__ wg,
                 __bf16* __restrict__ phiT,
                 __bf16* __restrict__ thetaT,
                 float* __restrict__ g) {
  __shared__ float Wp[Cc][Cc], Wt[Cc][Cc], Wg[Cc][Cc];
  const int b  = blockIdx.y;
  const int n0 = blockIdx.x * 64;
  const int t  = threadIdx.x;
  for (int i = t; i < Cc * Cc; i += 256) {
    Wp[i >> 6][i & 63] = wphi[i];
    Wt[i >> 6][i & 63] = wtheta[i];
    Wg[i >> 6][i & 63] = wg[i];
  }
  __syncthreads();
  const int o = t & 63;
  const float* xb = x + (size_t)b * Cc * Nn;
  for (int j = (t >> 6); j < 64; j += 4) {
    const int n = n0 + j;
    float ap = 0.f, at = 0.f, ag = 0.f;
#pragma unroll 8
    for (int c = 0; c < Cc; ++c) {
      float xv = xb[(size_t)c * Nn + n];
      ap += Wp[o][c] * xv;
      at += Wt[o][c] * xv;
      ag += Wg[o][c] * xv;
    }
    phiT[((size_t)b * Nn + n) * Cc + o]   = (__bf16)ap;
    thetaT[((size_t)b * Nn + n) * Cc + o] = (__bf16)at;
    g[((size_t)b * Cc + o) * Nn + n]      = ag;
  }
}

// ---------------------------------------------------------------------------
// Phase B: per-column (m) online max & sum of exp over all rows n.
// Each wave owns TWO 16-column tiles (theta fragments hoisted); running stats
// kept independently per lane-half, merged with one shuffle after the loop.
// ---------------------------------------------------------------------------
__global__ __launch_bounds__(64)
void colstats_kernel(const __bf16* __restrict__ phiT,
                     const __bf16* __restrict__ thetaT,
                     float* __restrict__ colmax,
                     float* __restrict__ colsum) {
  const int b    = blockIdx.y;
  const int lane = threadIdx.x & 31;
  const int wave = threadIdx.x >> 5;
  const int m0   = (blockIdx.x * 2 + wave) * 32;           // 250 jobs x 32 cols
  const int hl   = lane >> 4;
  const int l15  = lane & 15;
  const __bf16* phiB = phiT + (size_t)b * Nn * Cc;
  const __bf16* thB  = thetaT + (size_t)b * Nn * Cc;

  // hoisted theta B-fragments for both m-tiles (lane: N=m, K=c chunks)
  v16bf tb[2][2];
#pragma unroll
  for (int mt = 0; mt < 2; ++mt) {
    const __bf16* trow = thB + (size_t)(m0 + mt * 16 + l15) * Cc;
    tb[mt][0] = load16(trow + hl * 16);
    tb[mt][1] = load16(trow + 32 + hl * 16);
  }

  float runmax[2] = {-1e30f, -1e30f};
  float runsum[2] = {0.f, 0.f};
  const float scale = 0.125f;                              // 1/sqrt(64)

  for (int n0 = 0; n0 < Nn; n0 += 16) {
    const __bf16* prow = phiB + (size_t)(n0 + l15) * Cc;
    __builtin_prefetch(prow + 8 * Cc, 0, 1);               // stream phi ahead
    v16bf a0 = load16(prow + hl * 16);
    v16bf a1 = load16(prow + 32 + hl * 16);
#pragma unroll
    for (int mt = 0; mt < 2; ++mt) {
      v8f acc = {};
      acc = wmma_bf16(a0, tb[mt][0], acc);
      acc = wmma_bf16(a1, tb[mt][1], acc);
      // lane-half owns rows n0 + hl*8 + r of column m0 + mt*16 + l15
      float v[8], tmax = -1e30f;
#pragma unroll
      for (int r = 0; r < 8; ++r) { v[r] = acc[r] * scale; tmax = fmaxf(tmax, v[r]); }
      float nm = fmaxf(runmax[mt], tmax);
      float s = 0.f;
#pragma unroll
      for (int r = 0; r < 8; ++r) s += __expf(v[r] - nm);
      runsum[mt] = runsum[mt] * __expf(runmax[mt] - nm) + s;
      runmax[mt] = nm;
    }
  }
  // merge the two lane-halves (rows hl*8..hl*8+7 mod 16) once
#pragma unroll
  for (int mt = 0; mt < 2; ++mt) {
    float om = __shfl_xor(runmax[mt], 16, 32);
    float os = __shfl_xor(runsum[mt], 16, 32);
    float nm = fmaxf(runmax[mt], om);
    float tot = runsum[mt] * __expf(runmax[mt] - nm) + os * __expf(om - nm);
    if (lane < 16) {
      colmax[(size_t)b * Nn + m0 + mt * 16 + l15] = nm;
      colsum[(size_t)b * Nn + m0 + mt * 16 + l15] = tot;
    }
  }
}

// ---------------------------------------------------------------------------
// Phase B2: ghat[b][c][m] = bf16( g[b][c][m] / denom[b][m] )
// ---------------------------------------------------------------------------
__global__ __launch_bounds__(256)
void scaleg_kernel(const float* __restrict__ g,
                   const float* __restrict__ denom,
                   __bf16* __restrict__ ghat) {
  int i = blockIdx.x * 256 + threadIdx.x;
  if (i >= Bb * Cc * Nn) return;
  int n = i % Nn;
  int b = i / (Cc * Nn);
  ghat[i] = (__bf16)(g[i] / denom[(size_t)b * Nn + n]);
}

// ---------------------------------------------------------------------------
// Phase C: out_att[n][c] = sum_m exp(sim[n,m]-colmax[m]) * ghat[c][m]
// Each wave owns TWO 16-row n-tiles (phi fragments hoisted; theta/ghat
// fragments shared across both). m consumed in chunks of 32 (bf16 K).
// E tiles staged via LDS (same-wave DS is in-order) for D->A layout change.
// 16 WMMAs per iteration.
// ---------------------------------------------------------------------------
__global__ __launch_bounds__(64)
void attout_kernel(const __bf16* __restrict__ phiT,
                   const __bf16* __restrict__ thetaT,
                   const __bf16* __restrict__ ghat,
                   const float* __restrict__ colmax,
                   float* __restrict__ attout) {
  __shared__ float Es[2][2][16][34];                       // [wave][ntile][n][m]
  const int b    = blockIdx.y;
  const int lane = threadIdx.x & 31;
  const int wave = threadIdx.x >> 5;
  const int n0   = (blockIdx.x * 2 + wave) * 32;           // 250 jobs x 32 rows
  const int hl   = lane >> 4;
  const int l15  = lane & 15;
  const __bf16* phiB = phiT + (size_t)b * Nn * Cc;
  const __bf16* thB  = thetaT + (size_t)b * Nn * Cc;
  const __bf16* gB   = ghat + (size_t)b * Cc * Nn;
  const float*  cmB  = colmax + (size_t)b * Nn;
  const float scale = 0.125f;

  // hoisted phi A-fragments for both n-tiles
  v16bf a[2][2];
#pragma unroll
  for (int nt = 0; nt < 2; ++nt) {
    const __bf16* prow = phiB + (size_t)(n0 + nt * 16 + l15) * Cc;
    a[nt][0] = load16(prow + hl * 16);
    a[nt][1] = load16(prow + 32 + hl * 16);
  }

  v8f o[2][4];
#pragma unroll
  for (int nt = 0; nt < 2; ++nt)
#pragma unroll
    for (int t = 0; t < 4; ++t) o[nt][t] = (v8f){};

  for (int m0 = 0; m0 < Nn; m0 += 32) {
#pragma unroll
    for (int mt = 0; mt < 2; ++mt) {
      const int mm = m0 + mt * 16;
      const __bf16* trow = thB + (size_t)(mm + l15) * Cc;
      __builtin_prefetch(trow + 16 * Cc, 0, 1);            // stream theta ahead
      v16bf tb0 = load16(trow + hl * 16);
      v16bf tb1 = load16(trow + 32 + hl * 16);
      const float cm = cmB[mm + l15];                      // column max
#pragma unroll
      for (int nt = 0; nt < 2; ++nt) {
        v8f acc = {};
        acc = wmma_bf16(a[nt][0], tb0, acc);
        acc = wmma_bf16(a[nt][1], tb1, acc);
#pragma unroll
        for (int r = 0; r < 8; ++r)
          Es[wave][nt][hl * 8 + r][mt * 16 + l15] = __expf(acc[r] * scale - cm);
      }
    }
    // D-layout -> A-layout through LDS: lane row n=l15, elems {hl*16+i, hl*16+8+i}
    v16bf ea[2];
#pragma unroll
    for (int nt = 0; nt < 2; ++nt)
#pragma unroll
      for (int i = 0; i < 8; ++i) {
        ea[nt][i]     = (__bf16)Es[wave][nt][l15][hl * 16 + i];
        ea[nt][i + 8] = (__bf16)Es[wave][nt][l15][hl * 16 + 8 + i];
      }
    // ghat B-fragments shared by both n-tiles: lane N=c, K = m0 + hl*16 + 0..15
#pragma unroll
    for (int t = 0; t < 4; ++t) {
      const __bf16* grow = gB + (size_t)(t * 16 + l15) * Nn + m0 + hl * 16;
      v16bf gb = load16(grow);
#pragma unroll
      for (int nt = 0; nt < 2; ++nt) o[nt][t] = wmma_bf16(ea[nt], gb, o[nt][t]);
    }
  }
  // write attout [B][N][C] f32
  float* ao = attout + (size_t)b * Nn * Cc;
#pragma unroll
  for (int nt = 0; nt < 2; ++nt)
#pragma unroll
    for (int t = 0; t < 4; ++t)
#pragma unroll
      for (int r = 0; r < 8; ++r)
        ao[(size_t)(n0 + nt * 16 + hl * 8 + r) * Cc + t * 16 + l15] = o[nt][t][r];
}

// ---------------------------------------------------------------------------
// Phase D: final = gamma * (W_mask @ att) + x     (exact f32)
// ---------------------------------------------------------------------------
__global__ __launch_bounds__(256)
void mask_kernel(const float* __restrict__ att,            // [B][N][C]
                 const float* __restrict__ wmask,
                 const float* __restrict__ x,
                 const float* __restrict__ gamma,
                 float* __restrict__ out) {
  __shared__ float Wm[Cc][Cc + 1];
  __shared__ float Ar[4][Cc];
  const int b  = blockIdx.y;
  const int n0 = blockIdx.x * 4;
  const int t  = threadIdx.x;
  for (int i = t; i < Cc * Cc; i += 256) Wm[i >> 6][i & 63] = wmask[i];
  const int j = t >> 6, o = t & 63;
  Ar[j][o] = att[((size_t)b * Nn + n0 + j) * Cc + o];
  __syncthreads();
  float acc = 0.f;
#pragma unroll 8
  for (int c = 0; c < Cc; ++c) acc += Wm[o][c] * Ar[j][c];
  const size_t xi = ((size_t)b * Cc + o) * Nn + n0 + j;
  out[xi] = gamma[0] * acc + x[xi];
}

// ---------------------------------------------------------------------------
extern "C" void kernel_launch(void* const* d_in, const int* in_sizes, int n_in,
                              void* d_out, int out_size, void* d_ws, size_t ws_size,
                              hipStream_t stream) {
  const float* x      = (const float*)d_in[0];
  const float* wphi   = (const float*)d_in[1];
  const float* wtheta = (const float*)d_in[2];
  const float* wg     = (const float*)d_in[3];
  const float* wmask  = (const float*)d_in[4];
  const float* gamma  = (const float*)d_in[5];
  float* out = (float*)d_out;

  // workspace carve (all buffers fully overwritten before use)
  size_t off = 0;
  auto carve = [&](size_t bytes) -> void* {
    void* p = (char*)d_ws + off;
    off = (off + bytes + 255) & ~(size_t)255;
    return p;
  };
  __bf16* phiT   = (__bf16*)carve((size_t)Bb * Nn * Cc * 2);   // [B][N][C]
  __bf16* thetaT = (__bf16*)carve((size_t)Bb * Nn * Cc * 2);   // [B][N][C]
  float*  g      = (float*) carve((size_t)Bb * Cc * Nn * 4);   // [B][C][N]
  __bf16* ghat   = (__bf16*)carve((size_t)Bb * Cc * Nn * 2);   // [B][C][N]
  float*  cmax   = (float*) carve((size_t)Bb * Nn * 4);
  float*  csum   = (float*) carve((size_t)Bb * Nn * 4);
  float*  attO   = (float*) carve((size_t)Bb * Nn * Cc * 4);   // [B][N][C]
  (void)ws_size; (void)n_in; (void)in_sizes; (void)out_size;

  proj_kernel<<<dim3(Nn / 64, Bb), 256, 0, stream>>>(x, wphi, wtheta, wg,
                                                     phiT, thetaT, g);
  colstats_kernel<<<dim3(125, Bb), 64, 0, stream>>>(phiT, thetaT, cmax, csum);
  scaleg_kernel<<<(Bb * Cc * Nn + 255) / 256, 256, 0, stream>>>(g, csum, ghat);
  attout_kernel<<<dim3(125, Bb), 64, 0, stream>>>(phiT, thetaT, ghat, cmax, attO);
  mask_kernel<<<dim3(Nn / 4, Bb), 256, 0, stream>>>(attO, wmask, x, gamma, out);
}
